// Model_3710851744528
// MI455X (gfx1250) — compile-verified, matrix operand order checked by reading, not run on hardware
//
#include <hip/hip_runtime.h>
#include <math.h>

// ---------------------------------------------------------------------------
// FEDformer forward for MI455X (gfx1250, wave32, WMMA bf16 -> f32)
// ---------------------------------------------------------------------------
#define BB 32
#define SEQ 512
#define DMODEL 512
#define NHEAD 8
#define EDIM 64
#define DFF 2048
#define NMODES 64
#define CIN 21
#define TFEAT 4
#define DECLEN 512
#define PREDL 256
#define MROWS (BB*SEQ)          // 16384 rows for all big GEMMs

typedef __bf16 bf16_t;
typedef __attribute__((ext_vector_type(16))) __bf16 v16bf;
typedef __attribute__((ext_vector_type(8)))  __bf16 v8bf;
typedef __attribute__((ext_vector_type(4)))  __bf16 v4bf;
typedef __attribute__((ext_vector_type(8)))  float  v8f;

// ---------------------------------------------------------------------------
// WMMA GEMM: C[M,N] = act(alpha * A[M,K] @ B[K,N] + bias)
// Block tile 128x128 (8 waves, 4Mx2N), wave tile 32x64 (2x4 WMMA tiles).
// Register-pipelined: tile k+1 global loads overlap with tile k WMMA work.
// M%128==0, N%128==0, K%32==0.
// ---------------------------------------------------------------------------
__global__ void __launch_bounds__(256)
fed_gemm_wmma(const float* __restrict__ A, const float* __restrict__ Bm,
              const float* __restrict__ bias, float* __restrict__ C,
              int M, int N, int K, float alpha, int act)
{
    __shared__ __bf16 As[128][40];   // A tile rows, 80B row stride (16B aligned)
    __shared__ __bf16 Bs[128][40];   // B tile transposed: [n][k]
    const int tid  = threadIdx.x;
    const int bm   = blockIdx.x * 128;
    const int bn   = blockIdx.y * 128;
    const int wave = tid >> 5;
    const int lane = tid & 31;
    const int wm   = (wave & 3) * 32;   // 4 waves along M
    const int wn   = (wave >> 2) * 64;  // 2 waves along N
    const int lm   = lane & 15;
    const int lk   = lane >> 4;         // half-lane selector

    // staging coordinates
    const int ar0 = tid >> 3;           // A row   (+32*i)
    const int ac0 = (tid & 7) << 2;     // A col4
    const int bn0 = tid & 127;          // B col (n)
    const int bk0 = (tid >> 7) << 2;    // B k base (+8*i)

    // pointer-bump addressing (no per-iteration 64-bit rebuild)
    const float* aP = A + (size_t)(bm + ar0) * K + ac0;
    const float* bP = Bm + (size_t)bk0 * N + (bn + bn0);
    const size_t aOff1 = (size_t)32 * K;
    const size_t aOff2 = (size_t)64 * K;
    const size_t aOff3 = (size_t)96 * K;

    float4 a4[4];
    float  bb[4][4];

    auto load_tiles = [&]() {
        a4[0] = *(const float4*)(aP);
        a4[1] = *(const float4*)(aP + aOff1);
        a4[2] = *(const float4*)(aP + aOff2);
        a4[3] = *(const float4*)(aP + aOff3);
        #pragma unroll
        for (int i = 0; i < 4; ++i) {
            const float* bpp = bP + (size_t)(8 * i) * N;
            bb[i][0] = bpp[0];
            bb[i][1] = bpp[(size_t)N];
            bb[i][2] = bpp[(size_t)2 * N];
            bb[i][3] = bpp[(size_t)3 * N];
        }
        aP += 32;
        bP += (size_t)32 * N;
    };
    auto stage_tiles = [&]() {
        #pragma unroll
        for (int i = 0; i < 4; ++i) {
            v4bf p;
            p[0] = (__bf16)a4[i].x; p[1] = (__bf16)a4[i].y;
            p[2] = (__bf16)a4[i].z; p[3] = (__bf16)a4[i].w;
            *(v4bf*)&As[ar0 + 32 * i][ac0] = p;
        }
        #pragma unroll
        for (int i = 0; i < 4; ++i) {
            v4bf p;
            p[0] = (__bf16)bb[i][0]; p[1] = (__bf16)bb[i][1];
            p[2] = (__bf16)bb[i][2]; p[3] = (__bf16)bb[i][3];
            *(v4bf*)&Bs[bn0][bk0 + 8 * i] = p;
        }
    };

    v8f acc[2][4] = {};

    load_tiles();                        // prologue: tile 0 in registers
    for (int k0 = 0; k0 < K; k0 += 32) {
        stage_tiles();                   // registers -> LDS (bf16 packed)
        __syncthreads();
        if (k0 + 32 < K) load_tiles();   // next tile overlaps WMMA below

        // fragments: ISA A-layout puts K=lk*8..+7 and 16+lk*8..+7 contiguous
        v16bf af[2], bfv[4];
        #pragma unroll
        for (int t = 0; t < 2; ++t) {
            int row = wm + t * 16 + lm;
            v8bf lo = *(const v8bf*)&As[row][lk * 8];
            v8bf hi = *(const v8bf*)&As[row][16 + lk * 8];
            af[t] = __builtin_shufflevector(lo, hi, 0,1,2,3,4,5,6,7,8,9,10,11,12,13,14,15);
        }
        #pragma unroll
        for (int t = 0; t < 4; ++t) {
            int col = wn + t * 16 + lm;
            v8bf lo = *(const v8bf*)&Bs[col][lk * 8];
            v8bf hi = *(const v8bf*)&Bs[col][16 + lk * 8];
            bfv[t] = __builtin_shufflevector(lo, hi, 0,1,2,3,4,5,6,7,8,9,10,11,12,13,14,15);
        }
        #pragma unroll
        for (int i = 0; i < 2; ++i)
            #pragma unroll
            for (int j = 0; j < 4; ++j)
                acc[i][j] = __builtin_amdgcn_wmma_f32_16x16x32_bf16(
                    false, af[i], false, bfv[j], (short)0, acc[i][j], false, false);
        __syncthreads();
    }

    #pragma unroll
    for (int i = 0; i < 2; ++i) {
        #pragma unroll
        for (int j = 0; j < 4; ++j) {
            int col = bn + wn + j * 16 + lm;
            float bv = bias ? bias[col] : 0.0f;
            #pragma unroll
            for (int r = 0; r < 8; ++r) {
                int row = bm + wm + i * 16 + (lk << 3) + r;
                float v = acc[i][j][r] * alpha + bv;
                if (act == 1) v = 0.5f * v * (1.0f + erff(v * 0.70710678118f));
                C[(size_t)row * N + col] = v;
            }
        }
    }
}

// ---------------------------------------------------------------------------
// Per-batch 32x32-tiled transpose: Out[b,c,r] = In[b,r,c]
// ---------------------------------------------------------------------------
__global__ void __launch_bounds__(256)
fed_transpose(const float* __restrict__ In, float* __restrict__ Out, int R, int C)
{
    __shared__ float tile[32][33];
    int bc = blockIdx.x, br = blockIdx.y, b = blockIdx.z;
    int tx = threadIdx.x & 31, ty = threadIdx.x >> 5;
    const float* ip = In + (size_t)b * R * C;
    float* op = Out + (size_t)b * R * C;
    #pragma unroll
    for (int i = 0; i < 4; ++i) {
        int r = br * 32 + ty + i * 8, c = bc * 32 + tx;
        tile[ty + i * 8][tx] = ip[(size_t)r * C + c];
    }
    __syncthreads();
    #pragma unroll
    for (int i = 0; i < 4; ++i) {
        int c = bc * 32 + ty + i * 8, r = br * 32 + tx;
        op[(size_t)c * R + r] = tile[tx][ty + i * 8];
    }
}

// ---------------------------------------------------------------------------
// DFT bases. rfft: [L=512, 128] cols = (Re m<64 | Im m<64), X=x@Basis.
// irfft: [128, 512] folds 1/L, weight 2 for m>=1, drops Im of DC.
// ---------------------------------------------------------------------------
__global__ void fed_fill_rfft(float* Rb)
{
    int idx = blockIdx.x * 256 + threadIdx.x;
    if (idx >= 512 * 128) return;
    int mcol = idx & 127, l = idx >> 7;
    int m = mcol & 63;
    int p = (m * l) & 511;
    float ang = 6.28318530718f * (float)p / 512.0f;
    Rb[idx] = (mcol < 64) ? cosf(ang) : -sinf(ang);
}
__global__ void fed_fill_irfft(float* Ib)
{
    int idx = blockIdx.x * 256 + threadIdx.x;
    if (idx >= 128 * 512) return;
    int n = idx & 511, mrow = idx >> 9;
    int m = mrow & 63;
    float cm = (m == 0) ? 1.0f : 2.0f;
    int p = (m * n) & 511;
    float ang = 6.28318530718f * (float)p / 512.0f;
    float v;
    if (mrow < 64) v = cm * cosf(ang);
    else           v = (m == 0) ? 0.0f : -cm * sinf(ang);
    Ib[idx] = v / 512.0f;
}

// ---------------------------------------------------------------------------
// Complex per-mode mixing: Y[b,h,o,m] = sum_e X[b,h,e,m] * W[h,e,o,m]
// Xft/Yft layout: rows (b,h,e) x cols [Re 0..63 | Im 0..63]
// ---------------------------------------------------------------------------
__global__ void __launch_bounds__(64)
fed_mode_mix(const float* __restrict__ Xft, const float* __restrict__ W,
             float* __restrict__ Yft)
{
    int blk = blockIdx.x;
    int m = blk & 63, h = (blk >> 6) & 7, b = blk >> 9;
    __shared__ float xr[64], xi[64];
    int e = threadIdx.x;
    size_t rb = ((size_t)(b * 8 + h) * 64 + e) * 128;
    xr[e] = Xft[rb + m];
    xi[e] = Xft[rb + 64 + m];
    __syncthreads();
    int o = threadIdx.x;
    float ar = 0.f, ai = 0.f;
    for (int ee = 0; ee < 64; ++ee) {
        size_t wi = ((((size_t)h * 64 + ee) * 64 + o) * 64 + m) * 2;
        float wr = W[wi], wim = W[wi + 1];
        ar += xr[ee] * wr - xi[ee] * wim;
        ai += xr[ee] * wim + xi[ee] * wr;
    }
    size_t orow = ((size_t)(b * 8 + h) * 64 + o) * 128;
    Yft[orow + m] = ar;
    Yft[orow + 64 + m] = ai;
}

__device__ inline float2 fed_ctanh(float re, float im)
{
    float a = 2.f * re, c = 2.f * im;
    a = fminf(fmaxf(a, -30.f), 30.f);
    float d = coshf(a) + cosf(c);
    return make_float2(sinhf(a) / d, sinf(c) / d);
}

// xqk[b,h,x,y] = ctanh( sum_e q[b,h,e,x] * k[b,h,e,y] )  (complex)
__global__ void __launch_bounds__(256)
fed_cross_qk(const float* __restrict__ Qft, const float* __restrict__ Kft,
             float* __restrict__ XQK)
{
    int h = blockIdx.x & 7, b = blockIdx.x >> 3;
    __shared__ float qr[16][64], qi[16][64], kr[16][64], ki[16][64];
    float accr[16], acci[16];
    #pragma unroll
    for (int i = 0; i < 16; ++i) { accr[i] = 0.f; acci[i] = 0.f; }
    int tid = threadIdx.x;
    for (int ec = 0; ec < 64; ec += 16) {
        for (int i = 0; i < 4; ++i) {
            int idx = i * 256 + tid;
            int e = idx >> 6, x = idx & 63;
            size_t row = ((size_t)(b * 8 + h) * 64 + ec + e) * 128;
            qr[e][x] = Qft[row + x]; qi[e][x] = Qft[row + 64 + x];
            kr[e][x] = Kft[row + x]; ki[e][x] = Kft[row + 64 + x];
        }
        __syncthreads();
        #pragma unroll
        for (int i = 0; i < 16; ++i) {
            int p = i * 256 + tid;
            int x = p >> 6, y = p & 63;
            float ar = accr[i], ai = acci[i];
            #pragma unroll
            for (int e = 0; e < 16; ++e) {
                float a_r = qr[e][x], a_i = qi[e][x];
                float b_r = kr[e][y], b_i = ki[e][y];
                ar += a_r * b_r - a_i * b_i;
                ai += a_r * b_i + a_i * b_r;
            }
            accr[i] = ar; acci[i] = ai;
        }
        __syncthreads();
    }
    for (int i = 0; i < 16; ++i) {
        int p = i * 256 + tid;
        int x = p >> 6, y = p & 63;
        float2 t = fed_ctanh(accr[i], acci[i]);
        size_t o = (((size_t)(b * 8 + h) * 64 + x) * 64 + y) * 2;
        XQK[o] = t.x; XQK[o + 1] = t.y;
    }
}

// xqkv[b,h,e,x] = sum_y xqk[b,h,x,y] * k[b,h,e,y]  (complex), ft layout out
__global__ void __launch_bounds__(256)
fed_cross_av(const float* __restrict__ XQK, const float* __restrict__ Kft,
             float* __restrict__ Vft)
{
    int h = blockIdx.x & 7, b = blockIdx.x >> 3;
    __shared__ float sr[64][16], si[64][16], tr[64][16], ti[64][16];
    float accr[16], acci[16];
    #pragma unroll
    for (int i = 0; i < 16; ++i) { accr[i] = 0.f; acci[i] = 0.f; }
    int tid = threadIdx.x;
    for (int yc = 0; yc < 64; yc += 16) {
        for (int i = 0; i < 4; ++i) {
            int idx = i * 256 + tid;
            int r = idx >> 4, yl = idx & 15;
            size_t qo = (((size_t)(b * 8 + h) * 64 + r) * 64 + yc + yl) * 2;
            sr[r][yl] = XQK[qo]; si[r][yl] = XQK[qo + 1];
            size_t krow = ((size_t)(b * 8 + h) * 64 + r) * 128;
            tr[r][yl] = Kft[krow + yc + yl];
            ti[r][yl] = Kft[krow + 64 + yc + yl];
        }
        __syncthreads();
        #pragma unroll
        for (int i = 0; i < 16; ++i) {
            int p = i * 256 + tid;
            int e = p >> 6, x = p & 63;
            float ar = accr[i], ai = acci[i];
            #pragma unroll
            for (int yl = 0; yl < 16; ++yl) {
                float a_r = sr[x][yl], a_i = si[x][yl];
                float b_r = tr[e][yl], b_i = ti[e][yl];
                ar += a_r * b_r - a_i * b_i;
                ai += a_r * b_i + a_i * b_r;
            }
            accr[i] = ar; acci[i] = ai;
        }
        __syncthreads();
    }
    for (int i = 0; i < 16; ++i) {
        int p = i * 256 + tid;
        int e = p >> 6, x = p & 63;
        size_t row = ((size_t)(b * 8 + h) * 64 + e) * 128;
        Vft[row + x] = accr[i];
        Vft[row + 64 + x] = acci[i];
    }
}

// ---------------------------------------------------------------------------
// series_decomp on (X [+ Y]): S = v - movavg25(v), Mn = movavg25(v)
// ---------------------------------------------------------------------------
__global__ void fed_decomp(const float* __restrict__ X, const float* __restrict__ Y,
                           float* __restrict__ S, float* __restrict__ Mn,
                           int Bn, int L, int C)
{
    size_t idx = (size_t)blockIdx.x * 256 + threadIdx.x;
    size_t n = (size_t)Bn * L * C;
    if (idx >= n) return;
    int c = idx % C;
    int l = (idx / C) % L;
    int b = idx / ((size_t)C * L);
    float s = 0.f;
    #pragma unroll
    for (int j = -12; j <= 12; ++j) {
        int ll = l + j; ll = ll < 0 ? 0 : (ll >= L ? L - 1 : ll);
        size_t o = ((size_t)b * L + ll) * C + c;
        float v = X[o]; if (Y) v += Y[o];
        s += v;
    }
    float mean = s * (1.0f / 25.0f);
    size_t o0 = ((size_t)b * L + l) * C + c;
    float xv = X[o0]; if (Y) xv += Y[o0];
    if (S)  S[o0]  = xv - mean;
    if (Mn) Mn[o0] = mean;
}

__global__ void fed_add_inplace(float* __restrict__ D, const float* __restrict__ Sr, size_t n)
{
    size_t i = (size_t)blockIdx.x * 256 + threadIdx.x;
    if (i < n) D[i] += Sr[i];
}

// ---------------------------------------------------------------------------
// my_Layernorm part 1: LayerNorm over D per row  (safe in-place)
// ---------------------------------------------------------------------------
__global__ void __launch_bounds__(256)
fed_layernorm(const float* __restrict__ X, const float* __restrict__ g,
              const float* __restrict__ be, float* __restrict__ Y, int Dd)
{
    int row = blockIdx.x, tid = threadIdx.x;
    const float* xr = X + (size_t)row * Dd;
    float* yr = Y + (size_t)row * Dd;
    __shared__ float red[256];
    float s = 0.f;
    for (int d = tid; d < Dd; d += 256) s += xr[d];
    red[tid] = s; __syncthreads();
    for (int st = 128; st > 0; st >>= 1) { if (tid < st) red[tid] += red[tid + st]; __syncthreads(); }
    float mu = red[0] / Dd; __syncthreads();
    float s2 = 0.f;
    for (int d = tid; d < Dd; d += 256) { float v = xr[d] - mu; s2 += v * v; }
    red[tid] = s2; __syncthreads();
    for (int st = 128; st > 0; st >>= 1) { if (tid < st) red[tid] += red[tid + st]; __syncthreads(); }
    float inv = rsqrtf(red[0] / Dd + 1e-5f); __syncthreads();
    for (int d = tid; d < Dd; d += 256) yr[d] = (xr[d] - mu) * inv * g[d] + be[d];
}

// my_Layernorm part 2: subtract per-(b,d) mean over L (in place)
__global__ void __launch_bounds__(256)
fed_sub_time_mean(float* __restrict__ X, int L, int Dd)
{
    int b = blockIdx.x;
    for (int d = threadIdx.x; d < Dd; d += 256) {
        float s = 0.f;
        for (int l = 0; l < L; ++l) s += X[((size_t)b * L + l) * Dd + d];
        s /= (float)L;
        for (int l = 0; l < L; ++l) X[((size_t)b * L + l) * Dd + d] -= s;
    }
}

// ---------------------------------------------------------------------------
// Embedding: circular conv(21->512,k=3) + time-linear + sinusoidal PE
// ---------------------------------------------------------------------------
__global__ void __launch_bounds__(256)
fed_embed(const float* __restrict__ X, const float* __restrict__ XM,
          const float* __restrict__ convW, const float* __restrict__ timeW,
          float* __restrict__ Out, int L)
{
    int bl = blockIdx.x;
    int l = bl % L, b = bl / L;
    __shared__ float win[63];
    __shared__ float tm[4];
    int tid = threadIdx.x;
    if (tid < 63) {
        int kk = tid / 21, c = tid % 21;
        int ll = (l + kk - 1 + L) % L;
        win[tid] = X[((size_t)b * L + ll) * 21 + c];
    }
    if (tid >= 64 && tid < 68) tm[tid - 64] = XM[((size_t)b * L + l) * 4 + (tid - 64)];
    __syncthreads();
    for (int d = tid; d < 512; d += 256) {
        float acc = 0.f;
        for (int c = 0; c < 21; ++c)
            #pragma unroll
            for (int kk = 0; kk < 3; ++kk)
                acc += win[kk * 21 + c] * convW[((size_t)d * 21 + c) * 3 + kk];
        #pragma unroll
        for (int t = 0; t < 4; ++t) acc += tm[t] * timeW[(size_t)t * 512 + d];
        int i2 = d & ~1;
        float div = expf(-(float)i2 * (9.210340372f / 512.0f));
        float ang = (float)l * div;
        acc += (d & 1) ? cosf(ang) : sinf(ang);
        Out[((size_t)b * L + l) * 512 + d] = acc;
    }
}

// trend conv: circular conv(512->21,k=3), accumulated into Trend
__global__ void __launch_bounds__(64)
fed_trend_conv(const float* __restrict__ T, const float* __restrict__ W,
               float* __restrict__ Trend, int L)
{
    int bl = blockIdx.x;
    int l = bl % L, b = bl / L;
    __shared__ float win[3][512];
    int tid = threadIdx.x;
    for (int kk = 0; kk < 3; ++kk) {
        int ll = (l + kk - 1 + L) % L;
        for (int c = tid; c < 512; c += 64)
            win[kk][c] = T[((size_t)b * L + ll) * 512 + c];
    }
    __syncthreads();
    if (tid < 21) {
        float acc = 0.f;
        for (int c = 0; c < 512; ++c)
            #pragma unroll
            for (int kk = 0; kk < 3; ++kk)
                acc += win[kk][c] * W[((size_t)tid * 512 + c) * 3 + kk];
        Trend[((size_t)b * L + l) * 21 + tid] += acc;
    }
}

// per-(b,c) time mean of x_enc
__global__ void fed_time_mean21(const float* __restrict__ X, float* __restrict__ M, int L)
{
    int idx = blockIdx.x * 64 + threadIdx.x;
    if (idx >= BB * CIN) return;
    int c = idx % CIN, b = idx / CIN;
    float s = 0.f;
    for (int l = 0; l < L; ++l) s += X[((size_t)b * L + l) * CIN + c];
    M[idx] = s / (float)L;
}

// build trend init + decoder seasonal init
__global__ void fed_build_init(const float* __restrict__ trendI, const float* __restrict__ seasE,
                               const float* __restrict__ meanE, float* __restrict__ Trend,
                               float* __restrict__ SeasD)
{
    size_t idx = (size_t)blockIdx.x * 256 + threadIdx.x;
    if (idx >= (size_t)BB * DECLEN * CIN) return;
    int c = idx % CIN;
    int l = (idx / CIN) % DECLEN;
    int b = idx / ((size_t)CIN * DECLEN);
    if (l < 256) {
        size_t src = ((size_t)b * SEQ + (256 + l)) * CIN + c;
        Trend[idx] = trendI[src];
        SeasD[idx] = seasE[src];
    } else {
        Trend[idx] = meanE[b * CIN + c];
        SeasD[idx] = 0.f;
    }
}

// final: out[b,l,c] = trend[b,256+l,c] + Xn[b,256+l,:] @ projW[:,c] + projb[c]
__global__ void __launch_bounds__(64)
fed_final(const float* __restrict__ Xn, const float* __restrict__ Trend,
          const float* __restrict__ Wp, const float* __restrict__ bp,
          float* __restrict__ Out)
{
    int idx = blockIdx.x;
    int lp = idx % PREDL, b = idx / PREDL;
    int ld = 256 + lp;
    __shared__ float xr[512];
    for (int d = threadIdx.x; d < 512; d += 64)
        xr[d] = Xn[((size_t)b * DECLEN + ld) * 512 + d];
    __syncthreads();
    if (threadIdx.x < CIN) {
        int c = threadIdx.x;
        float acc = bp[c];
        for (int d = 0; d < 512; ++d) acc += xr[d] * Wp[(size_t)d * CIN + c];
        acc += Trend[((size_t)b * DECLEN + ld) * CIN + c];
        Out[((size_t)b * PREDL + lp) * CIN + c] = acc;
    }
}

// ---------------------------------------------------------------------------
// Host helpers
// ---------------------------------------------------------------------------
static inline void h_gemm(hipStream_t st, const float* A, const float* B, const float* bias,
                          float* C, int M, int N, int K, float alpha = 1.f, int act = 0)
{
    dim3 g(M / 128, N / 128, 1);
    fed_gemm_wmma<<<g, 256, 0, st>>>(A, B, bias, C, M, N, K, alpha, act);
}
static inline void h_transpose(hipStream_t st, const float* In, float* Out, int Bn, int R, int C)
{
    dim3 g(C / 32, R / 32, Bn);
    fed_transpose<<<g, 256, 0, st>>>(In, Out, R, C);
}
static inline void h_decomp(hipStream_t st, const float* X, const float* Y,
                            float* S, float* Mn, int Bn, int L, int C)
{
    size_t n = (size_t)Bn * L * C;
    fed_decomp<<<(unsigned)((n + 255) / 256), 256, 0, st>>>(X, Y, S, Mn, Bn, L, C);
}

// fourier self-attention block (K/V projections are dead code in reference)
static void h_fattn(hipStream_t st, const float* x,
                    const float* Wq, const float* bq, const float* fw,
                    const float* Wo, const float* bo,
                    float* sQ, float* sT, float* Xft, float* Yft,
                    const float* RB, const float* IB, float* out)
{
    h_gemm(st, x, Wq, bq, sQ, MROWS, DMODEL, DMODEL);
    h_transpose(st, sQ, sT, BB, SEQ, DMODEL);             // rows (b,h,e) x L
    h_gemm(st, sT, RB, nullptr, Xft, MROWS, 128, SEQ);    // rFFT modes
    fed_mode_mix<<<BB * NHEAD * NMODES, 64, 0, st>>>(Xft, fw, Yft);
    h_gemm(st, Yft, IB, nullptr, sQ, MROWS, SEQ, 128);    // irFFT
    h_transpose(st, sQ, sT, BB, DMODEL, SEQ);             // back to (b,l,d)
    h_gemm(st, sT, Wo, bo, out, MROWS, DMODEL, DMODEL);
}

// ---------------------------------------------------------------------------
extern "C" void kernel_launch(void* const* d_in, const int* in_sizes, int n_in,
                              void* d_out, int out_size, void* d_ws, size_t ws_size,
                              hipStream_t stream)
{
    (void)in_sizes; (void)n_in; (void)out_size; (void)ws_size;
    const float* x_enc   = (const float*)d_in[0];
    const float* xm_enc  = (const float*)d_in[1];
    const float* xm_dec  = (const float*)d_in[3];
    const float* e_convW = (const float*)d_in[4];
    const float* e_timeW = (const float*)d_in[5];
    const float* d_convW = (const float*)d_in[6];
    const float* d_timeW = (const float*)d_in[7];
    const float* fw_enc  = (const float*)d_in[8];    // complex64 interleaved
    const float* fw_dec  = (const float*)d_in[9];
    const float* fw_crs  = (const float*)d_in[10];
    // enc layers: base 11 and 21: [Wq bq Wk bk Wv bv Wo bo] W1 W2
    const float* enc_g = (const float*)d_in[31];
    const float* enc_b = (const float*)d_in[32];
    // dec layer: self 33..40, cross 41..48, W1 49, W2 50, trendW 51
    const float* dec_g = (const float*)d_in[52];
    const float* dec_b = (const float*)d_in[53];
    const float* projW = (const float*)d_in[54];
    const float* projb = (const float*)d_in[55];
    float* out = (float*)d_out;

    // ---- workspace layout (floats) ----
    float* W = (float*)d_ws;
    size_t off = 0;
    const size_t SBD = (size_t)BB * SEQ * DMODEL;
    #define ALLOC(name, n) float* name = W + off; off += (n)
    ALLOC(RB,    (size_t)512 * 128);
    ALLOC(IB,    (size_t)128 * 512);
    ALLOC(encx,  SBD);
    ALLOC(decx,  SBD);
    ALLOC(xb,    SBD);
    ALLOC(sQ,    SBD);
    ALLOC(sT,    SBD);
    ALLOC(sS,    SBD);
    ALLOC(aout,  SBD);
    ALLOC(ffn,   (size_t)BB * SEQ * DFF);
    ALLOC(tsum,  SBD);
    ALLOC(tbuf,  SBD);
    ALLOC(Xft,   (size_t)MROWS * 128);
    ALLOC(Yft,   (size_t)MROWS * 128);
    ALLOC(Kft,   (size_t)MROWS * 128);
    ALLOC(Vft,   (size_t)MROWS * 128);
    ALLOC(XQK,   (size_t)BB * NHEAD * 64 * 64 * 2);
    ALLOC(trend, (size_t)BB * DECLEN * CIN);
    ALLOC(trendI,(size_t)BB * SEQ * CIN);
    ALLOC(seasE, (size_t)BB * SEQ * CIN);
    ALLOC(seasD, (size_t)BB * DECLEN * CIN);
    ALLOC(meanE, (size_t)BB * CIN);
    #undef ALLOC

    // ---- DFT bases ----
    fed_fill_rfft<<<(512 * 128 + 255) / 256, 256, 0, stream>>>(RB);
    fed_fill_irfft<<<(128 * 512 + 255) / 256, 256, 0, stream>>>(IB);

    // ---- stage 0: input decomposition & inits ----
    h_decomp(stream, x_enc, nullptr, seasE, trendI, BB, SEQ, CIN);
    fed_time_mean21<<<(BB * CIN + 63) / 64, 64, 0, stream>>>(x_enc, meanE, SEQ);
    {
        size_t n = (size_t)BB * DECLEN * CIN;
        fed_build_init<<<(unsigned)((n + 255) / 256), 256, 0, stream>>>(trendI, seasE, meanE, trend, seasD);
    }
    fed_embed<<<BB * SEQ, 256, 0, stream>>>(x_enc, xm_enc, e_convW, e_timeW, encx, SEQ);
    fed_embed<<<BB * DECLEN, 256, 0, stream>>>(seasD, xm_dec, d_convW, d_timeW, decx, DECLEN);

    // ---- encoder (2 layers, shared fw_enc) ----
    for (int layer = 0; layer < 2; ++layer) {
        int base = 11 + layer * 10;
        const float* Wq = (const float*)d_in[base + 0];
        const float* bq = (const float*)d_in[base + 1];
        const float* Wo = (const float*)d_in[base + 6];
        const float* bo = (const float*)d_in[base + 7];
        const float* W1 = (const float*)d_in[base + 8];
        const float* W2 = (const float*)d_in[base + 9];
        h_fattn(stream, encx, Wq, bq, fw_enc, Wo, bo, sQ, sT, Xft, Yft, RB, IB, aout);
        h_decomp(stream, encx, aout, xb, nullptr, BB, SEQ, DMODEL);
        h_gemm(stream, xb, W1, nullptr, ffn, MROWS, DFF, DMODEL, 1.f, 1);   // GELU
        h_gemm(stream, ffn, W2, nullptr, aout, MROWS, DMODEL, DFF);
        h_decomp(stream, xb, aout, encx, nullptr, BB, SEQ, DMODEL);
    }
    fed_layernorm<<<MROWS, 256, 0, stream>>>(encx, enc_g, enc_b, encx, DMODEL);
    fed_sub_time_mean<<<BB, 256, 0, stream>>>(encx, SEQ, DMODEL);

    // ---- decoder (1 layer) ----
    {
        const float* sWq = (const float*)d_in[33]; const float* sbq = (const float*)d_in[34];
        const float* sWo = (const float*)d_in[39]; const float* sbo = (const float*)d_in[40];
        const float* cWq = (const float*)d_in[41]; const float* cbq = (const float*)d_in[42];
        const float* cWk = (const float*)d_in[43]; const float* cbk = (const float*)d_in[44];
        const float* cWo = (const float*)d_in[47]; const float* cbo = (const float*)d_in[48];
        const float* W1  = (const float*)d_in[49]; const float* W2  = (const float*)d_in[50];
        const float* tW  = (const float*)d_in[51];
        const size_t nBD = SBD;

        // self freq-attn
        h_fattn(stream, decx, sWq, sbq, fw_dec, sWo, sbo, sQ, sT, Xft, Yft, RB, IB, aout);
        h_decomp(stream, decx, aout, xb, tsum, BB, DECLEN, DMODEL);         // x, t1

        // cross freq-attn: q from xb, kv from encx
        h_gemm(stream, xb, cWq, cbq, sQ, MROWS, DMODEL, DMODEL);
        h_transpose(stream, sQ, sT, BB, DECLEN, DMODEL);
        h_gemm(stream, sT, RB, nullptr, Xft, MROWS, 128, DECLEN);
        h_gemm(stream, encx, cWk, cbk, sQ, MROWS, DMODEL, DMODEL);
        h_transpose(stream, sQ, sS, BB, SEQ, DMODEL);
        h_gemm(stream, sS, RB, nullptr, Kft, MROWS, 128, SEQ);
        fed_cross_qk<<<BB * NHEAD, 256, 0, stream>>>(Xft, Kft, XQK);
        fed_cross_av<<<BB * NHEAD, 256, 0, stream>>>(XQK, Kft, Vft);
        fed_mode_mix<<<BB * NHEAD * NMODES, 64, 0, stream>>>(Vft, fw_crs, Yft);
        h_gemm(stream, Yft, IB, nullptr, sQ, MROWS, DECLEN, 128,
               1.0f / (512.0f * 512.0f), 0);                                 // /D/D
        h_transpose(stream, sQ, sT, BB, DMODEL, DECLEN);
        h_gemm(stream, sT, cWo, cbo, aout, MROWS, DMODEL, DMODEL);

        h_decomp(stream, xb, aout, decx, tbuf, BB, DECLEN, DMODEL);          // x, t2
        fed_add_inplace<<<(unsigned)((nBD + 255) / 256), 256, 0, stream>>>(tsum, tbuf, nBD);

        // FFN
        h_gemm(stream, decx, W1, nullptr, ffn, MROWS, DFF, DMODEL, 1.f, 1);  // GELU
        h_gemm(stream, ffn, W2, nullptr, aout, MROWS, DMODEL, DFF);
        h_decomp(stream, decx, aout, xb, tbuf, BB, DECLEN, DMODEL);          // x, t3
        fed_add_inplace<<<(unsigned)((nBD + 255) / 256), 256, 0, stream>>>(tsum, tbuf, nBD);

        // residual trend projection
        fed_trend_conv<<<BB * DECLEN, 64, 0, stream>>>(tsum, tW, trend, DECLEN);

        // final norm
        fed_layernorm<<<MROWS, 256, 0, stream>>>(xb, dec_g, dec_b, decx, DMODEL);
        fed_sub_time_mean<<<BB, 256, 0, stream>>>(decx, DECLEN, DMODEL);
    }

    // ---- projection + trend add, last PRED steps ----
    fed_final<<<BB * PREDL, 64, 0, stream>>>(decx, trend, projW, projb, out);
}
